// FourRegionAttention_66571993088976
// MI455X (gfx1250) — matrix operand order repaired and checked by reading, not compiled, not run on hardware
//
#include <hip/hip_runtime.h>

typedef __attribute__((ext_vector_type(16))) _Float16 v16h;
typedef __attribute__((ext_vector_type(8)))  _Float16 v8h;
typedef __attribute__((ext_vector_type(4)))  _Float16 v4h;
typedef __attribute__((ext_vector_type(8)))  float    v8f;

#define N_B    8
#define N_POS  1024
#define N_DIM  128
#define N_HEAD 8

// ---------------------------------------------------------------------------
// 1) LayerNorm: one wave per row (128 elems, 4 per lane).
//    Writes xn as f16 [B*N,128] (GEMM A operand) and f32 transposed [B,128,N]
//    (coalesced source for the per-channel integral image).
// ---------------------------------------------------------------------------
__global__ __launch_bounds__(256) void ln_kernel(
    const float* __restrict__ x, const float* __restrict__ g,
    const float* __restrict__ beta,
    _Float16* __restrict__ xn_h, float* __restrict__ xn_t)
{
  int wave = threadIdx.x >> 5, lane = threadIdx.x & 31;
  int row  = blockIdx.x * 8 + wave;            // 0..8191
  const float4 xv = *(const float4*)(x + (size_t)row * 128 + lane * 4);
  float s  = xv.x + xv.y + xv.z + xv.w;
  float ss = xv.x*xv.x + xv.y*xv.y + xv.z*xv.z + xv.w*xv.w;
  #pragma unroll
  for (int off = 16; off > 0; off >>= 1) {
    s  += __shfl_xor(s,  off, 32);
    ss += __shfl_xor(ss, off, 32);
  }
  float mean = s * (1.0f / 128.0f);
  float var  = ss * (1.0f / 128.0f) - mean * mean;
  float rstd = rsqrtf(var + 1e-5f);
  float4 gv = *(const float4*)(g    + lane * 4);
  float4 bv = *(const float4*)(beta + lane * 4);
  float o0 = (xv.x - mean) * rstd * gv.x + bv.x;
  float o1 = (xv.y - mean) * rstd * gv.y + bv.y;
  float o2 = (xv.z - mean) * rstd * gv.z + bv.z;
  float o3 = (xv.w - mean) * rstd * gv.w + bv.w;
  v4h oh; oh[0] = (_Float16)o0; oh[1] = (_Float16)o1;
  oh[2] = (_Float16)o2; oh[3] = (_Float16)o3;
  *(v4h*)(xn_h + (size_t)row * 128 + lane * 4) = oh;
  int b = row >> 10, n = row & 1023;
  float* tp = xn_t + (size_t)b * 128 * 1024 + n;
  tp[(lane * 4 + 0) * 1024] = o0;
  tp[(lane * 4 + 1) * 1024] = o1;
  tp[(lane * 4 + 2) * 1024] = o2;
  tp[(lane * 4 + 3) * 1024] = o3;
}

// ---------------------------------------------------------------------------
// 2) Region means: one wave per (b,d). Lane = column w; 32 rows in registers.
//    Lane-scan (shfl_up) for row prefix, sequential for column prefix, then
//    the 4 quadrant means from the inclusive integral image. f16 output
//    [b, n, r, 128] = GEMM A operand for Wk/Wv.
// ---------------------------------------------------------------------------
__global__ __launch_bounds__(256) void region_kernel(
    const float* __restrict__ xn_t, _Float16* __restrict__ regions)
{
  int wave = threadIdx.x >> 5, lane = threadIdx.x & 31;
  int wg = blockIdx.x * 8 + wave;              // 0..1023
  int b = wg >> 7, d = wg & 127;
  const float* src = xn_t + ((size_t)b * 128 + d) * 1024;
  float v[32];
  #pragma unroll
  for (int h = 0; h < 32; ++h) v[h] = src[h * 32 + lane];
  #pragma unroll
  for (int h = 0; h < 32; ++h) {
    #pragma unroll
    for (int off = 1; off < 32; off <<= 1) {
      float t = __shfl_up(v[h], off, 32);
      if (lane >= off) v[h] += t;
    }
  }
  #pragma unroll
  for (int h = 1; h < 32; ++h) v[h] += v[h - 1];
  float colT   = v[31];                           // P(31, w)
  float S      = __shfl(v[31], 31, 32);           // P(31,31)
  float colTm1 = __shfl_up(v[31], 1, 32);         // P(31, w-1)
  if (lane == 0) colTm1 = 0.0f;
  float prevRow = 0.0f, prevRowT = 0.0f, prevRowM1 = 0.0f;
  #pragma unroll
  for (int h = 0; h < 32; ++h) {
    float cur   = v[h];
    float rowT  = __shfl(cur, 31, 32);            // P(h,31)
    float curM1 = __shfl_up(cur, 1, 32);          // P(h,w-1)
    if (lane == 0) curM1 = 0.0f;
    float r1 = cur;
    float r2 = rowT - curM1;
    float r3 = colT - prevRow;
    float r4 = S - prevRowT - colTm1 + prevRowM1;
    float hp1 = (float)(h + 1),   hr = (float)(32 - h);
    float wp1 = (float)(lane + 1), wr = (float)(32 - lane);
    size_t base = (((size_t)b * 1024 + h * 32 + lane) * 4) * 128 + d;
    regions[base      ] = (_Float16)(r1 / (hp1 * wp1));
    regions[base + 128] = (_Float16)(r2 / (hp1 * wr));
    regions[base + 256] = (_Float16)(r3 / (hr * wp1));
    regions[base + 384] = (_Float16)(r4 / (hr * wr));
    prevRow = cur; prevRowT = rowT; prevRowM1 = curM1;
  }
}

// ---------------------------------------------------------------------------
// 3) Fused-weight precompute (f32, tiny): Wfq[h] = Wqkv[:,h] @ Wq,
//    Wbig[h*128+i,j] = sum_m Wo[i,m]*Wout[h*128+m,j], and the fused biases.
// ---------------------------------------------------------------------------
__global__ __launch_bounds__(256) void fuse_wq_kernel(
    const float* __restrict__ Wqkv, const float* __restrict__ Wq,
    float* __restrict__ Wfq)
{
  int idx = blockIdx.x * 256 + threadIdx.x;     // h*16384 + k*128 + j
  int h = idx >> 14, k = (idx >> 7) & 127, j = idx & 127;
  float acc = 0.0f;
  for (int i = 0; i < 128; ++i)
    acc += Wqkv[k * 3072 + h * 128 + i] * Wq[i * 128 + j];
  Wfq[idx] = acc;
}

__global__ __launch_bounds__(256) void fuse_wbig_kernel(
    const float* __restrict__ Wo, const float* __restrict__ Wout,
    float* __restrict__ Wbig)
{
  int idx = blockIdx.x * 256 + threadIdx.x;     // h*16384 + i*128 + j
  int h = idx >> 14, i = (idx >> 7) & 127, j = idx & 127;
  float acc = 0.0f;
  for (int m = 0; m < 128; ++m)
    acc += Wo[i * 128 + m] * Wout[(h * 128 + m) * 128 + j];
  Wbig[idx] = acc;
}

__global__ __launch_bounds__(256) void fuse_bias_kernel(
    const float* __restrict__ bqkv, const float* __restrict__ Wq,
    const float* __restrict__ bo, const float* __restrict__ Wout,
    const float* __restrict__ bout,
    float* __restrict__ bq, float* __restrict__ bfin)
{
  int idx = blockIdx.x * 256 + threadIdx.x;
  if (idx < 1024) {                             // bq[h][j]
    int h = idx >> 7, j = idx & 127;
    float acc = 0.0f;
    for (int i = 0; i < 128; ++i) acc += bqkv[h * 128 + i] * Wq[i * 128 + j];
    bq[idx] = acc;
  } else if (idx < 1152) {                      // bfin[j]
    int j = idx - 1024;
    float acc = 0.0f;
    for (int t = 0; t < 1024; ++t) acc += bo[t & 127] * Wout[t * 128 + j];
    bfin[j] = acc + bout[j];
  }
}

// ---------------------------------------------------------------------------
// 4) Pack f32 weight [nb][K][128] into per-lane WMMA B fragments (f16).
//    frag index = ((kstep*8 + ntile)*32 + lane)*16 + j
//    maps to element k = kstep*32 + (lane/16)*16 + j, n = ntile*16 + lane%16.
// ---------------------------------------------------------------------------
__global__ __launch_bounds__(256) void pack_b_kernel(
    const float* __restrict__ src, _Float16* __restrict__ dst, int K, int nb)
{
  int idx = blockIdx.x * 256 + threadIdx.x;
  int per = K * 128;
  if (idx >= nb * per) return;
  int batch = idx / per, t = idx % per;
  int kstep = t >> 12;
  int r = t & 4095;
  int ntile = r >> 9;
  int r2 = r & 511;
  int lane = r2 >> 4, j = r2 & 15;
  int k = kstep * 32 + (lane >> 4) * 16 + j;
  int n = ntile * 16 + (lane & 15);
  dst[idx] = (_Float16)src[(size_t)batch * per + k * 128 + n];
}

// ---------------------------------------------------------------------------
// 5) WMMA GEMM, LDS-staged: C[z] = A[z] (f16 [M,K]) * Bpack[z] + bias, f32.
//    N fixed at 128: 8 waves/block = one N-tile each; 64 rows per block.
//    The 64x32 A tile is shared by all 8 waves -> stage it in LDS once
//    (cooperative b128 fill, double-buffered, row stride padded to 40 halves
//    so half-wave ds_load_b128 starts tile all 64 banks conflict-free).
//    Software pipeline: issue next tile's global load FIRST, compute the
//    4 WMMAs, then commit the prefetch to LDS -> s_wait_loadcnt sinks below
//    the matrix math and HBM/L2 latency overlaps compute.
// ---------------------------------------------------------------------------
#define TILE_STRIDE 40                      // halves per LDS row (32 + 8 pad)

__global__ __launch_bounds__(256) void gemm_wmma_kernel(
    const _Float16* __restrict__ A, const _Float16* __restrict__ Bp,
    const float* __restrict__ bias, float* __restrict__ C,
    int M, int K, int a_div, int a_stride, int b_mod, int b_stride,
    int bias_mod, int c_stride)
{
  __shared__ _Float16 tile[2][64 * TILE_STRIDE];
  int tid  = threadIdx.x;
  int lane = tid & 31;
  int wave = tid >> 5;                        // ntile
  int z = blockIdx.z;
  const _Float16* Az = A + (size_t)(z / a_div) * a_stride;
  const _Float16* Bz = Bp + (size_t)(z % b_mod) * b_stride;
  float* Cz = C + (size_t)z * c_stride;
  int m0 = blockIdx.x * 64;
  int mrow = lane & 15;
  int g = lane >> 4;
  int ksteps = K >> 5;

  // Cooperative A-tile fill mapping: thread -> (row, 8-half segment).
  int crow = tid >> 2, cseg = tid & 3;
  const _Float16* Asrc = Az + (size_t)(m0 + crow) * K + cseg * 8;
  int ldst_off = crow * TILE_STRIDE + cseg * 8;

  // Preload K-step 0.
  *(v8h*)(&tile[0][ldst_off]) = *(const v8h*)(Asrc);
  __syncthreads();

  v8f acc[4] = {};
  union AU { v16h v; v8h h[2]; };
  for (int ks = 0; ks < ksteps; ++ks) {
    int cur = ks & 1;
    bool have_pf = (ks + 1 < ksteps);
    v8h pf;
    if (have_pf)                              // issue prefetch load early...
      pf = *(const v8h*)(Asrc + (ks + 1) * 32);
    v16h bfrag = *(const v16h*)(Bz + ((((size_t)ks * 8 + wave) * 32) + lane) * 16);
    const _Float16* tb = &tile[cur][0];
    #pragma unroll
    for (int mt = 0; mt < 4; ++mt) {
      const _Float16* p = tb + (mt * 16 + mrow) * TILE_STRIDE + g * 16;
      AU a;
      a.h[0] = *(const v8h*)(p);
      a.h[1] = *(const v8h*)(p + 8);
      acc[mt] = __builtin_amdgcn_wmma_f32_16x16x32_f16(
          false, a.v, false, bfrag, (short)0, acc[mt], false, false);
    }
    if (have_pf)                              // ...commit it after the WMMAs
      *(v8h*)(&tile[cur ^ 1][ldst_off]) = pf;
    __syncthreads();
  }

  int n = wave * 16 + (lane & 15);
  float bval = bias ? bias[(z % bias_mod) * 128 + n] : 0.0f;
  #pragma unroll
  for (int mt = 0; mt < 4; ++mt) {
    #pragma unroll
    for (int r = 0; r < 8; ++r) {
      int m = m0 + mt * 16 + r + 8 * g;
      Cz[(size_t)m * 128 + n] = acc[mt][r] + bval;
    }
  }
}

// ---------------------------------------------------------------------------
// 6) Attention: one wave per (b,n). Lane holds 4 channels. 8 heads x 4 regions
//    dot products via wave reduce, softmax over r=4, weighted v-sum; f16
//    head-concat output -> A operand of the final fused GEMM.
// ---------------------------------------------------------------------------
__global__ __launch_bounds__(256) void attn_kernel(
    const float* __restrict__ qh, const float* __restrict__ kb,
    const float* __restrict__ vb, _Float16* __restrict__ aggcat)
{
  int wave = threadIdx.x >> 5, lane = threadIdx.x & 31;
  int wg = blockIdx.x * 8 + wave;               // 0..8191 = b*1024+n
  int b = wg >> 10, n = wg & 1023;
  size_t kvb = ((size_t)wg * 4) * 128 + lane * 4;
  float4 k0 = *(const float4*)(kb + kvb);
  float4 k1 = *(const float4*)(kb + kvb + 128);
  float4 k2 = *(const float4*)(kb + kvb + 256);
  float4 k3 = *(const float4*)(kb + kvb + 384);
  float4 v0 = *(const float4*)(vb + kvb);
  float4 v1 = *(const float4*)(vb + kvb + 128);
  float4 v2 = *(const float4*)(vb + kvb + 256);
  float4 v3 = *(const float4*)(vb + kvb + 384);
  for (int h = 0; h < N_HEAD; ++h) {
    const float4 q = *(const float4*)(
        qh + (((size_t)b * 8 + h) * 1024 + n) * 128 + lane * 4);
    float s0 = q.x*k0.x + q.y*k0.y + q.z*k0.z + q.w*k0.w;
    float s1 = q.x*k1.x + q.y*k1.y + q.z*k1.z + q.w*k1.w;
    float s2 = q.x*k2.x + q.y*k2.y + q.z*k2.z + q.w*k2.w;
    float s3 = q.x*k3.x + q.y*k3.y + q.z*k3.z + q.w*k3.w;
    #pragma unroll
    for (int off = 16; off > 0; off >>= 1) {
      s0 += __shfl_xor(s0, off, 32);
      s1 += __shfl_xor(s1, off, 32);
      s2 += __shfl_xor(s2, off, 32);
      s3 += __shfl_xor(s3, off, 32);
    }
    s0 *= 0.25f; s1 *= 0.25f; s2 *= 0.25f; s3 *= 0.25f;   // (dim//HEADS)^-0.5
    float mx = fmaxf(fmaxf(s0, s1), fmaxf(s2, s3));
    float e0 = expf(s0 - mx), e1 = expf(s1 - mx);
    float e2 = expf(s2 - mx), e3 = expf(s3 - mx);
    float inv = 1.0f / (e0 + e1 + e2 + e3);
    e0 *= inv; e1 *= inv; e2 *= inv; e3 *= inv;
    float a0 = e0*v0.x + e1*v1.x + e2*v2.x + e3*v3.x;
    float a1 = e0*v0.y + e1*v1.y + e2*v2.y + e3*v3.y;
    float a2 = e0*v0.z + e1*v1.z + e2*v2.z + e3*v3.z;
    float a3 = e0*v0.w + e1*v1.w + e2*v2.w + e3*v3.w;
    v4h o; o[0] = (_Float16)a0; o[1] = (_Float16)a1;
    o[2] = (_Float16)a2; o[3] = (_Float16)a3;
    *(v4h*)(aggcat + (size_t)wg * 1024 + h * 128 + lane * 4) = o;
  }
}

// ---------------------------------------------------------------------------
extern "C" void kernel_launch(void* const* d_in, const int* in_sizes, int n_in,
                              void* d_out, int out_size, void* d_ws, size_t ws_size,
                              hipStream_t stream) {
  (void)in_sizes; (void)n_in; (void)out_size; (void)ws_size;
  const float* x    = (const float*)d_in[0];
  const float* ln_g = (const float*)d_in[1];
  const float* ln_b = (const float*)d_in[2];
  const float* Wqkv = (const float*)d_in[3];
  const float* bqkv = (const float*)d_in[4];
  const float* Wq   = (const float*)d_in[5];
  const float* Wk   = (const float*)d_in[6];
  const float* Wv   = (const float*)d_in[7];
  const float* Wo   = (const float*)d_in[8];
  const float* bo   = (const float*)d_in[9];
  const float* Wout = (const float*)d_in[10];
  const float* bout = (const float*)d_in[11];
  float* out = (float*)d_out;

  char* ws = (char*)d_ws;
  size_t off = 0;
  auto alloc = [&](size_t bytes) {
    size_t o = off;
    off = (off + bytes + 255) & ~(size_t)255;
    return (void*)(ws + o);
  };
  float*    xn_t    = (float*)   alloc((size_t)8 * 128 * 1024 * 4);   //  4 MB
  _Float16* xn_h    = (_Float16*)alloc((size_t)8 * 1024 * 128 * 2);   //  2 MB
  _Float16* regions = (_Float16*)alloc((size_t)8 * 1024 * 4 * 128 * 2); // 8 MB
  float*    Wfq     = (float*)   alloc((size_t)8 * 128 * 128 * 4);
  float*    bq      = (float*)   alloc((size_t)8 * 128 * 4);
  float*    Wbig    = (float*)   alloc((size_t)1024 * 128 * 4);
  float*    bfin    = (float*)   alloc((size_t)128 * 4);
  _Float16* packq   = (_Float16*)alloc((size_t)8 * 128 * 128 * 2);
  _Float16* packk   = (_Float16*)alloc((size_t)128 * 128 * 2);
  _Float16* packv   = (_Float16*)alloc((size_t)128 * 128 * 2);
  _Float16* packbig = (_Float16*)alloc((size_t)1024 * 128 * 2);
  float*    qhb     = (float*)   alloc((size_t)64 * 1024 * 128 * 4);  // 32 MB
  float*    kbuf    = (float*)   alloc((size_t)32768 * 128 * 4);      // 16 MB
  float*    vbuf    = (float*)   alloc((size_t)32768 * 128 * 4);      // 16 MB
  _Float16* aggcat  = (_Float16*)alloc((size_t)8192 * 1024 * 2);      // 16 MB

  // Stage 1: LayerNorm (+ transpose, + f16 copy)
  ln_kernel<<<1024, 256, 0, stream>>>(x, ln_g, ln_b, xn_h, xn_t);
  // Stage 2: region means (integral image per (b,d))
  region_kernel<<<128, 256, 0, stream>>>(xn_t, regions);
  // Stage 3: fused weights
  fuse_wq_kernel<<<512, 256, 0, stream>>>(Wqkv, Wq, Wfq);
  fuse_wbig_kernel<<<512, 256, 0, stream>>>(Wo, Wout, Wbig);
  fuse_bias_kernel<<<5, 256, 0, stream>>>(bqkv, Wq, bo, Wout, bout, bq, bfin);
  // Stage 4: pack B operands into WMMA fragment layout
  pack_b_kernel<<<512, 256, 0, stream>>>(Wfq,  packq,   128, 8);
  pack_b_kernel<<<64,  256, 0, stream>>>(Wk,   packk,   128, 1);
  pack_b_kernel<<<64,  256, 0, stream>>>(Wv,   packv,   128, 1);
  pack_b_kernel<<<512, 256, 0, stream>>>(Wbig, packbig, 1024, 1);
  // Stage 5: WMMA GEMMs
  //   qh[b,h] = xn[b] @ Wfq[h] + bq[h]   (z = b*8+h)
  gemm_wmma_kernel<<<dim3(16, 1, 64), 256, 0, stream>>>(
      xn_h, packq, bq, qhb,
      1024, 128, /*a_div=*/8, /*a_stride=*/1024 * 128,
      /*b_mod=*/8, /*b_stride=*/128 * 128, /*bias_mod=*/8,
      /*c_stride=*/1024 * 128);
  //   k = regions @ Wk ; v = regions @ Wv
  gemm_wmma_kernel<<<dim3(512, 1, 1), 256, 0, stream>>>(
      regions, packk, nullptr, kbuf, 32768, 128, 1, 0, 1, 0, 1, 0);
  gemm_wmma_kernel<<<dim3(512, 1, 1), 256, 0, stream>>>(
      regions, packv, nullptr, vbuf, 32768, 128, 1, 0, 1, 0, 1, 0);
  // Stage 6: scores + softmax + weighted sum -> head-concat f16
  attn_kernel<<<1024, 256, 0, stream>>>(qhb, kbuf, vbuf, aggcat);
  // Stage 7: final fused projection: out = aggcat @ Wbig + bfin
  gemm_wmma_kernel<<<dim3(128, 1, 1), 256, 0, stream>>>(
      aggcat, packbig, bfin, out, 8192, 1024, 1, 0, 1, 0, 1, 0);
}
